// lstmPredictor_8323646620576
// MI455X (gfx1250) — compile-verified, matrix operand order checked by reading, not run on hardware
//
#include <hip/hip_runtime.h>
#include <hip/hip_bf16.h>
#include <stdint.h>

// ---------------------------------------------------------------------------
// 2-layer LSTM (T=512, B=64, I=256, H=1024) + FC(1024->1) for MI455X gfx1250.
//
//   * bf16 weights/activations, fp32 WMMA accumulate (v_wmma_f32_16x16x32_bf16).
//   * One kernel per (timestep, layer); graph-captured. Weights (~25 MB bf16)
//     stay resident in the 192 MB L2 across all 1024 steps.
//   * A-operand (activation) panel staged per block in LDS (padded pitch ->
//     conflict-free ds_load_b128); weights stream from L2 (zero reuse).
//   * K is a template parameter: both GEMM phases fully unroll into straight-
//     line code with explicit register double-buffering (loads for k+1 issued
//     before the WMMAs of k), enabling partial s_wait counts.
//   * Unconditional strided global_prefetch warms the weight stream.
//
// Workspace layout (bytes), total ~171 MB:
#define OFF_SEQB   ((size_t)0)              // T*B*I bf16      = 16,777,216
#define OFF_WIH0   ((size_t)16777216)       // 4096*256 bf16   =  2,097,152
#define OFF_WHH0   ((size_t)18874368)       // 4096*1024 bf16  =  8,388,608
#define OFF_WIH1   ((size_t)27262976)       // 4096*1024 bf16  =  8,388,608
#define OFF_WHH1   ((size_t)35651584)       // 4096*1024 bf16  =  8,388,608
#define OFF_HA     ((size_t)44040192)       // T*B*H bf16      = 67,108,864
#define OFF_HB     ((size_t)111149056)      // T*B*H bf16      = 67,108,864
#define OFF_ZEROH  ((size_t)178257920)      // B*H bf16 zeros  =    131,072
#define OFF_CA     ((size_t)178388992)      // B*H f32         =    262,144
#define OFF_CB     ((size_t)178651136)      // B*H f32         =    262,144
#define WS_NEEDED  ((size_t)178913280)

#define T_LEN 512
#define B_DIM 64
#define I_DIM 256
#define H_DIM 1024
#define LDS_PAD 8                            // elements; 16 B pad per row

typedef __bf16 bf16_t;
typedef bf16_t v16bf __attribute__((ext_vector_type(16)));
typedef float  v8f   __attribute__((ext_vector_type(8)));

struct Frag2x128 { uint4 lo; uint4 hi; };

__device__ __forceinline__ uint16_t f32_to_bf16_rne(float f) {
  uint32_t u = __float_as_uint(f);
  u += 0x7FFFu + ((u >> 16) & 1u);
  return (uint16_t)(u >> 16);
}
__device__ __forceinline__ float bf16_to_f32(uint16_t h) {
  return __uint_as_float(((uint32_t)h) << 16);
}
__device__ __forceinline__ float fast_sigmoid(float x) {
  return 1.0f / (1.0f + __expf(-x));
}
__device__ __forceinline__ float fast_tanh(float x) {
  return 1.0f - 2.0f / (__expf(2.0f * x) + 1.0f);   // saturates for |x| large
}

__device__ __forceinline__ Frag2x128 load_pair(const uint16_t* p) {
  Frag2x128 f;
  f.lo = *reinterpret_cast<const uint4*>(p);
  f.hi = *reinterpret_cast<const uint4*>(p + 16);
  return f;
}

// ---------------------------------------------------------------------------
// Fully-unrolled, register-double-buffered GEMM phase:
//   acc[g] += A(16xK from LDS panel) x B(Kx16 from W[g*1024+n0 .. +15][:]).
// Loads for step k+32 are issued before the WMMAs of step k.
template <int K>
__device__ __forceinline__ void gemm_phase(const uint16_t* __restrict__ W,
                                           int n0, int lane,
                                           const uint16_t* sA, v8f acc[4]) {
  const int kh = (lane >> 4) << 3;                     // 0 or 8
  const uint16_t* wrow[4];
#pragma unroll
  for (int g = 0; g < 4; ++g)
    wrow[g] = W + (size_t)(g * H_DIM + n0 + (lane & 15)) * K + kh;
  const uint16_t* arow = sA + (lane & 15) * (K + LDS_PAD) + kh;

  Frag2x128 af[2];
  Frag2x128 bf[2][4];

  af[0] = load_pair(arow);
#pragma unroll
  for (int g = 0; g < 4; ++g) bf[0][g] = load_pair(wrow[g]);

#pragma unroll
  for (int k0 = 0; k0 < K - 32; k0 += 32) {
    const int par = (k0 >> 5) & 1;
    const int np  = par ^ 1;
    const int nk  = k0 + 32;
    // Issue next-step fragment loads first (double buffer).
    af[np] = load_pair(arow + nk);
#pragma unroll
    for (int g = 0; g < 4; ++g) {
      bf[np][g] = load_pair(wrow[g] + nk);
      if ((nk & 63) == 0)                               // every 2nd k-step
        __builtin_prefetch(wrow[g] + nk + 256, 0, 3);   // 512 B ahead
    }
    // Consume current buffer.
    const v16bf a = __builtin_bit_cast(v16bf, af[par]);
#pragma unroll
    for (int g = 0; g < 4; ++g)
      acc[g] = __builtin_amdgcn_wmma_f32_16x16x32_bf16(
          false, a, false, __builtin_bit_cast(v16bf, bf[par][g]), (short)0,
          acc[g], false, false);
  }
  // Tail: consume the last prefetched buffer.
  const int lpar = ((K - 32) >> 5) & 1;
  const v16bf a = __builtin_bit_cast(v16bf, af[lpar]);
#pragma unroll
  for (int g = 0; g < 4; ++g)
    acc[g] = __builtin_amdgcn_wmma_f32_16x16x32_bf16(
        false, a, false, __builtin_bit_cast(v16bf, bf[lpar][g]), (short)0,
        acc[g], false, false);
}

// ---------------------------------------------------------------------------
// One LSTM timestep for one layer. Grid: 32 blocks x 256 threads = 256 waves.
// Wave w = blockIdx*8+wv: mt = w>>6 (uniform per block), nt = w&63.
template <int KIN>
__global__ __launch_bounds__(256)
void lstm_step_kernel(const uint16_t* __restrict__ x,    // [B, KIN] bf16
                      const uint16_t* __restrict__ Wih,  // [4096, KIN] bf16
                      const uint16_t* __restrict__ Whh,  // [4096, 1024] bf16
                      const float* __restrict__ bih,     // [4096]
                      const float* __restrict__ bhh,     // [4096]
                      const uint16_t* __restrict__ hprev,// [B, 1024] bf16
                      float* __restrict__ c,             // [B, 1024] f32 state
                      uint16_t* __restrict__ hnext) {    // [B, 1024] bf16
  __shared__ uint16_t sA[16 * (H_DIM + LDS_PAD)];        // 33,024 bytes

  const int tid  = threadIdx.x;
  const int lane = tid & 31;
  const int w    = blockIdx.x * 8 + (tid >> 5);
  const int m0   = (w >> 6) << 4;                        // batch-tile row base
  const int n0   = (w & 63) << 4;                        // hidden-tile base
  const int col  = n0 + (lane & 15);

  // Accumulators start at the (row-invariant) combined bias.
  v8f acc[4];
#pragma unroll
  for (int g = 0; g < 4; ++g) {
    const float bv = bih[g * H_DIM + col] + bhh[g * H_DIM + col];
#pragma unroll
    for (int p = 0; p < 8; ++p) acc[g][p] = bv;
  }

  // ---- Phase 1: x_t @ W_ih^T -------------------------------------------
  {
#pragma unroll
    for (int ci = tid; ci < 16 * (KIN >> 3); ci += 256) {
      const int r = ci / (KIN >> 3);
      const int j = (ci % (KIN >> 3)) << 3;
      *reinterpret_cast<uint4*>(&sA[r * (KIN + LDS_PAD) + j]) =
          *reinterpret_cast<const uint4*>(x + (size_t)(m0 + r) * KIN + j);
    }
    __syncthreads();
    gemm_phase<KIN>(Wih, n0, lane, sA, acc);
  }

  // ---- Phase 2: h_{t-1} @ W_hh^T ---------------------------------------
  {
    __syncthreads();                                     // done reading sA
#pragma unroll
    for (int ci = tid; ci < 16 * (H_DIM >> 3); ci += 256) {
      const int r = ci >> 7;                             // 128 chunks / row
      const int j = (ci & 127) << 3;
      *reinterpret_cast<uint4*>(&sA[r * (H_DIM + LDS_PAD) + j]) =
          *reinterpret_cast<const uint4*>(hprev + (size_t)(m0 + r) * H_DIM + j);
    }
    __syncthreads();
    gemm_phase<H_DIM>(Whh, n0, lane, sA, acc);
  }

  // ---- Fused LSTM cell update ------------------------------------------
  // C/D layout: VGPR p -> row m0 + p (+8 for lanes>=16), col = n0+(lane&15).
  const int rbase = m0 + ((lane >> 4) << 3);
#pragma unroll
  for (int p = 0; p < 8; ++p) {
    const size_t idx = (size_t)(rbase + p) * H_DIM + (size_t)col;
    const float iv = fast_sigmoid(acc[0][p]);
    const float fv = fast_sigmoid(acc[1][p]);
    const float gv = fast_tanh(acc[2][p]);
    const float ov = fast_sigmoid(acc[3][p]);
    const float cn = fv * c[idx] + iv * gv;
    c[idx] = cn;
    hnext[idx] = f32_to_bf16_rne(ov * fast_tanh(cn));
  }
}

// ---------------------------------------------------------------------------
__global__ __launch_bounds__(256)
void f32_to_bf16_kernel(const float* __restrict__ in, uint16_t* __restrict__ out,
                        size_t n) {
  size_t i = (size_t)blockIdx.x * blockDim.x + threadIdx.x;
  const size_t stride = (size_t)gridDim.x * blockDim.x;
  for (; i < n; i += stride) out[i] = f32_to_bf16_rne(in[i]);
}

__global__ __launch_bounds__(256)
void zero_u32_kernel(uint32_t* __restrict__ p, size_t n) {
  size_t i = (size_t)blockIdx.x * blockDim.x + threadIdx.x;
  const size_t stride = (size_t)gridDim.x * blockDim.x;
  for (; i < n; i += stride) p[i] = 0u;
}

// out[row] = relu(h2[row]) . W_fc + b_fc ; one wave per row.
__global__ __launch_bounds__(256)
void fc_head_kernel(const uint16_t* __restrict__ h,  // [nrows, 1024] bf16
                    const float* __restrict__ Wfc,   // [1024]
                    const float* __restrict__ bfc,   // [1]
                    float* __restrict__ out, int nrows) {
  const int row = blockIdx.x * 8 + (threadIdx.x >> 5);
  if (row >= nrows) return;
  const int lane = threadIdx.x & 31;
  const uint16_t* hr = h + (size_t)row * H_DIM;
  float s = 0.0f;
#pragma unroll 4
  for (int j = lane; j < H_DIM; j += 32) {
    float hv = bf16_to_f32(hr[j]);
    hv = hv > 0.0f ? hv : 0.0f;
    s += hv * Wfc[j];
  }
#pragma unroll
  for (int off = 16; off > 0; off >>= 1) s += __shfl_xor(s, off, 32);
  if (lane == 0) out[row] = s + bfc[0];
}

// ---------------------------------------------------------------------------
extern "C" void kernel_launch(void* const* d_in, const int* in_sizes, int n_in,
                              void* d_out, int out_size, void* d_ws, size_t ws_size,
                              hipStream_t stream) {
  const float* seq   = (const float*)d_in[0];
  const float* Wih0  = (const float*)d_in[1];
  const float* Whh0  = (const float*)d_in[2];
  const float* bih0  = (const float*)d_in[3];
  const float* bhh0  = (const float*)d_in[4];
  const float* Wih1  = (const float*)d_in[5];
  const float* Whh1  = (const float*)d_in[6];
  const float* bih1  = (const float*)d_in[7];
  const float* bhh1  = (const float*)d_in[8];
  const float* Wfc   = (const float*)d_in[9];
  const float* bfc   = (const float*)d_in[10];
  float* out = (float*)d_out;

  char* ws = (char*)d_ws;
  uint16_t* seqb  = (uint16_t*)(ws + OFF_SEQB);
  uint16_t* Wih0b = (uint16_t*)(ws + OFF_WIH0);
  uint16_t* Whh0b = (uint16_t*)(ws + OFF_WHH0);
  uint16_t* Wih1b = (uint16_t*)(ws + OFF_WIH1);
  uint16_t* Whh1b = (uint16_t*)(ws + OFF_WHH1);
  uint16_t* hA    = (uint16_t*)(ws + OFF_HA);     // layer-0 hidden, all T
  uint16_t* hB    = (uint16_t*)(ws + OFF_HB);     // layer-1 hidden, all T
  uint16_t* zeroH = (uint16_t*)(ws + OFF_ZEROH);  // h_{-1} = 0
  float*    cA    = (float*)(ws + OFF_CA);
  float*    cB    = (float*)(ws + OFF_CB);

  // bf16 conversions (once per launch; weights stay hot in the 192 MB L2).
  f32_to_bf16_kernel<<<2048, 256, 0, stream>>>(seq,  seqb,  (size_t)T_LEN * B_DIM * I_DIM);
  f32_to_bf16_kernel<<<1024, 256, 0, stream>>>(Wih0, Wih0b, (size_t)4 * H_DIM * I_DIM);
  f32_to_bf16_kernel<<<2048, 256, 0, stream>>>(Whh0, Whh0b, (size_t)4 * H_DIM * H_DIM);
  f32_to_bf16_kernel<<<2048, 256, 0, stream>>>(Wih1, Wih1b, (size_t)4 * H_DIM * H_DIM);
  f32_to_bf16_kernel<<<2048, 256, 0, stream>>>(Whh1, Whh1b, (size_t)4 * H_DIM * H_DIM);
  // Zero h_{-1} and both cell states (zeroH, cA, cB are contiguous in ws).
  zero_u32_kernel<<<160, 256, 0, stream>>>((uint32_t*)(ws + OFF_ZEROH),
                                           (size_t)(131072 + 2 * 262144) / 4);

  const size_t hstep = (size_t)B_DIM * H_DIM;  // 65536 elements per timestep
  for (int t = 0; t < T_LEN; ++t) {
    const uint16_t* hprevA = t ? (hA + (size_t)(t - 1) * hstep) : zeroH;
    lstm_step_kernel<I_DIM><<<32, 256, 0, stream>>>(
        seqb + (size_t)t * B_DIM * I_DIM,
        Wih0b, Whh0b, bih0, bhh0, hprevA, cA, hA + (size_t)t * hstep);

    const uint16_t* hprevB = t ? (hB + (size_t)(t - 1) * hstep) : zeroH;
    lstm_step_kernel<H_DIM><<<32, 256, 0, stream>>>(
        hA + (size_t)t * hstep,
        Wih1b, Whh1b, bih1, bhh1, hprevB, cB, hB + (size_t)t * hstep);
  }

  // FC head: 32768 rows, 8 rows per 256-thread block.
  fc_head_kernel<<<(T_LEN * B_DIM) / 8, 256, 0, stream>>>(hB, Wfc, bfc, out,
                                                          T_LEN * B_DIM);
  (void)in_sizes; (void)n_in; (void)out_size; (void)ws_size;
}
